// Attn_25409026523783
// MI455X (gfx1250) — matrix operand order repaired and checked by reading, compiled, not run on hardware
//
#include <hip/hip_runtime.h>

typedef __bf16 bf16_t;
typedef __attribute__((ext_vector_type(16))) __bf16 v16bf;
typedef __attribute__((ext_vector_type(8)))  __bf16 v8bf;
typedef __attribute__((ext_vector_type(4)))  __bf16 v4bf;
typedef __attribute__((ext_vector_type(8)))  float  v8f;

#define NB   64
#define NS   256
#define ND   2048
#define NH   16
#define NHD  128
#define NROWS (NB*NS)   // 16384
#define N3D  (3*ND)     // 6144

// ---------------- CDNA5 async global->LDS copy (ASYNCcnt-tracked) ----------------
// GLOBAL_LOAD_ASYNC_TO_LDS_B128: VDST = LDS byte address, VADDR = 64-bit global addr.
__device__ __forceinline__ void async_cp16(void* lds, const void* g) {
  unsigned l = (unsigned)(unsigned long long)
      (__attribute__((address_space(3))) char*)lds;             // LDS byte address
  unsigned long long a = (unsigned long long)g;
  asm volatile("global_load_async_to_lds_b128 %0, %1, off" :: "v"(l), "v"(a) : "memory");
}
__device__ __forceinline__ void wait_async0() {
  asm volatile("s_wait_asynccnt 0x0" ::: "memory");
}

// ---------------- WMMA fragment helpers (bf16, 16x16x32) ----------------
// A operand (16 rows x 32 K): lane m=lane&15, half=lane>>4.
//   vector elems 0..7  -> K = half*8 + i ;  elems 8..15 -> K = 16 + half*8 + (i-8)
template <typename PT>
__device__ __forceinline__ v16bf load_a_frag(PT tile, int ld) {
  const int lane = threadIdx.x & 31;
  const int m = lane & 15, h = lane >> 4;
  auto p = tile + m * ld + h * 8;
  v8bf lo = *(const v8bf*)(p);
  v8bf hi = *(const v8bf*)(p + 16);
  v16bf r;
#pragma unroll
  for (int i = 0; i < 8; ++i) { r[i] = lo[i]; r[i + 8] = hi[i]; }
  return r;
}
// B operand (32 K x 16 cols): lane n, elems 0..7 -> K = half*8 + i ; elems 8..15 -> K = 16 + half*8 + (i-8).
// tile points to row-major [N][K] data (columns of B).
template <typename PT>
__device__ __forceinline__ v16bf load_b_frag(PT tile, int ld) {
  const int lane = threadIdx.x & 31;
  const int n = lane & 15, h = lane >> 4;
  auto p = tile + n * ld + h * 8;
  v8bf lo = *(const v8bf*)(p);
  v8bf hi = *(const v8bf*)(p + 16);
  v16bf r;
#pragma unroll
  for (int i = 0; i < 8; ++i) { r[i] = lo[i]; r[i + 8] = hi[i]; }
  return r;
}
__device__ __forceinline__ v8f vzero8() {
  v8f z;
#pragma unroll
  for (int i = 0; i < 8; ++i) z[i] = 0.0f;
  return z;
}
#define WMMA_BF16(a, b, c) \
  __builtin_amdgcn_wmma_f32_16x16x32_bf16(false, (a), false, (b), (short)0, (c), false, false)

// ---------------- fp32 -> bf16 conversion ----------------
__global__ __launch_bounds__(256) void cvt_bf16_kernel(const float* __restrict__ s,
                                                       bf16_t* __restrict__ d, int n4) {
  int i = blockIdx.x * 256 + threadIdx.x;
  if (i < n4) {
    float4 f = ((const float4*)s)[i];
    v4bf o;
    o[0] = (bf16_t)f.x; o[1] = (bf16_t)f.y; o[2] = (bf16_t)f.z; o[3] = (bf16_t)f.w;
    ((v4bf*)d)[i] = o;
  }
}

// ---------------- block-tiled GEMM core: 128x128 tile, K-panels of 64 -------------
// 8 waves in 4(M) x 2(N); wave tile 32x64 -> acc[2][4]. LDS double-buffered via
// async-to-LDS copies. Padded LDS row stride 72 bf16 (=36 dwords, conflict-free).
#define BK   64
#define TLD  72
#define GEMM_SMEM (4 * 128 * TLD * 2)   // A[2] + B[2] panels = 73,728 B

// Each kernel instantiates the same loop; epilogues differ.
#define GEMM_PROLOGUE(NTILES_, AG_, BG_)                                         \
  extern __shared__ char smem[];                                                 \
  bf16_t* Abuf = (bf16_t*)smem;                                                  \
  bf16_t* Bbuf = Abuf + 2 * 128 * TLD;                                           \
  const int t = threadIdx.x;                                                     \
  const int lane = t & 31, wave = t >> 5;                                        \
  const int m = lane & 15, half = lane >> 4;                                     \
  const int wm = wave >> 1, wn = wave & 1;                                       \
  const int mtB = (blockIdx.x / (NTILES_)) * 128;                                \
  const int ntB = (blockIdx.x % (NTILES_)) * 128;                                \
  const bf16_t* Ag = (AG_) + (size_t)mtB * ND;                                   \
  const bf16_t* Bg = (BG_) + (size_t)ntB * ND;                                   \
  v8f acc[2][4];                                                                 \
  _Pragma("unroll") for (int f = 0; f < 2; ++f)                                  \
  _Pragma("unroll") for (int j = 0; j < 4; ++j) acc[f][j] = vzero8();            \
  auto stage = [&](int buf, int k0) {                                            \
    bf16_t* As = Abuf + buf * 128 * TLD;                                         \
    bf16_t* Bs = Bbuf + buf * 128 * TLD;                                         \
    _Pragma("unroll") for (int i = 0; i < 4; ++i) {                              \
      int c = t + i * 256; /* 0..1023: 128 rows x 8 chunks of 8 bf16 */          \
      int row = c >> 3, cc = c & 7;                                              \
      async_cp16(As + row * TLD + cc * 8, Ag + (size_t)row * ND + k0 + cc * 8);  \
      async_cp16(Bs + row * TLD + cc * 8, Bg + (size_t)row * ND + k0 + cc * 8);  \
    }                                                                            \
  };                                                                             \
  stage(0, 0);                                                                   \
  wait_async0();                                                                 \
  __syncthreads();                                                               \
  for (int kt = 0; kt < ND / BK; ++kt) {                                         \
    if (kt + 1 < ND / BK) stage((kt + 1) & 1, (kt + 1) * BK);                    \
    const bf16_t* As = Abuf + (kt & 1) * 128 * TLD + (wm * 32) * TLD;            \
    const bf16_t* Bs = Bbuf + (kt & 1) * 128 * TLD + (wn * 64) * TLD;            \
    _Pragma("unroll") for (int sub = 0; sub < 2; ++sub) {                        \
      v16bf a0 = load_a_frag(As + sub * 32, TLD);                                \
      v16bf a1 = load_a_frag(As + 16 * TLD + sub * 32, TLD);                     \
      _Pragma("unroll") for (int j = 0; j < 4; ++j) {                            \
        v16bf b = load_b_frag(Bs + (j * 16) * TLD + sub * 32, TLD);              \
        acc[0][j] = WMMA_BF16(a0, b, acc[0][j]);                                 \
        acc[1][j] = WMMA_BF16(a1, b, acc[1][j]);                                 \
      }                                                                          \
    }                                                                            \
    wait_async0();                                                               \
    __syncthreads();                                                             \
  }

// ---------------- QKV projection GEMM: scatter to Q/K/Vt head layouts ----------
__global__ __launch_bounds__(256) void qkv_gemm_kernel(
    const bf16_t* __restrict__ xb, const bf16_t* __restrict__ wb,
    bf16_t* __restrict__ q, bf16_t* __restrict__ k, bf16_t* __restrict__ vt) {
  GEMM_PROLOGUE(N3D / 128, xb, wb)

  // e = ntB + wn*64 + j*16 + n ; 64-wide span never crosses a head (HD=128) boundary
  int e0 = ntB + wn * 64;
  int which = e0 >> 11;          // 0=q 1=k 2=v
  int h = (e0 >> 7) & 15;
  int hd0 = e0 & 127;
  int b = mtB >> 8;              // 128-row span never crosses a batch (S=256) boundary
  int rowbase = (mtB & 255) + wm * 32;
  int bh = b * NH + h;
  if (which < 2) {
    bf16_t* dst = (which ? k : q) + (size_t)bh * NS * NHD;
#pragma unroll
    for (int f = 0; f < 2; ++f)
#pragma unroll
      for (int j = 0; j < 4; ++j)
#pragma unroll
        for (int r = 0; r < 8; ++r)
          dst[(size_t)(rowbase + f * 16 + r + 8 * half) * NHD + hd0 + j * 16 + m] =
              (bf16_t)acc[f][j][r];
  } else {
    bf16_t* dst = vt + (size_t)bh * NHD * NS;  // V stored transposed [HD][S]
#pragma unroll
    for (int f = 0; f < 2; ++f)
#pragma unroll
      for (int j = 0; j < 4; ++j)
#pragma unroll
        for (int r = 0; r < 8; ++r)
          dst[(size_t)(hd0 + j * 16 + m) * NS + rowbase + f * 16 + r + 8 * half] =
              (bf16_t)acc[f][j][r];
  }
}

// ---------------- output projection GEMM ----------------
__global__ __launch_bounds__(256) void out_gemm_kernel(
    const bf16_t* __restrict__ ctx, const bf16_t* __restrict__ wb,
    float* __restrict__ out) {
  GEMM_PROLOGUE(ND / 128, ctx, wb)

  float* dst = out + (size_t)(mtB + wm * 32) * ND + ntB + wn * 64;
#pragma unroll
  for (int f = 0; f < 2; ++f)
#pragma unroll
    for (int j = 0; j < 4; ++j)
#pragma unroll
      for (int r = 0; r < 8; ++r)
        dst[(size_t)(f * 16 + r + 8 * half) * ND + j * 16 + m] = acc[f][j][r];
}

// ---------------- causal attention per (b, h, 64-query block) ----------------
#define KLD 136   // 128 + 8 bf16 pad
#define VLD 264   // 256 + 8 bf16 pad
#define SLD 257   // 256 + 1 f32 pad
#define PLD 264   // 256 + 8 bf16 pad
#define ATTN_SMEM (NS*KLD*2 + NHD*VLD*2 + 4*16*SLD*4 + 4*16*PLD*2 + 4*16*4)

__global__ __launch_bounds__(128) void attn_kernel(
    const bf16_t* __restrict__ qws, const bf16_t* __restrict__ kws,
    const bf16_t* __restrict__ vtws, bf16_t* __restrict__ ctx) {
  extern __shared__ char smem[];
  bf16_t* Ks = (bf16_t*)smem;                 // [256][KLD]
  bf16_t* Vt = Ks + NS * KLD;                 // [128][VLD]
  float*  Sc = (float*)(Vt + NHD * VLD);      // 4 waves x [16][SLD]
  bf16_t* Pr = (bf16_t*)(Sc + 4 * 16 * SLD);  // 4 waves x [16][PLD]
  float*  Rs = (float*)(Pr + 4 * 16 * PLD);   // 4 waves x [16]

  const int bid = blockIdx.x;
  const int qblk = bid & 3;
  const int bh = bid >> 2;
  const int t = threadIdx.x;
  const int wave = t >> 5, lane = t & 31;
  const int m = lane & 15, half = lane >> 4;

  const bf16_t* kbase = kws + (size_t)bh * NS * NHD;
  const bf16_t* vbase = vtws + (size_t)bh * NHD * NS;
  const bf16_t* qbg   = qws + (size_t)bh * NS * NHD;

  // async stage: K [256][128] and V^T [128][256] into padded LDS
  for (int c = t; c < NS * (NHD / 8); c += 128) {
    int row = c >> 4, cc = c & 15;
    async_cp16(Ks + row * KLD + cc * 8, kbase + row * NHD + cc * 8);
  }
  for (int c = t; c < NHD * (NS / 8); c += 128) {
    int row = c >> 5, cc = c & 31;
    async_cp16(Vt + row * VLD + cc * 8, vbase + row * NS + cc * 8);
  }
  wait_async0();
  __syncthreads();

  const int qbase = qblk * 64 + wave * 16;
  float*  myS = Sc + wave * 16 * SLD;
  bf16_t* myP = Pr + wave * 16 * PLD;
  float*  myR = Rs + wave * 16;

  // Q fragments for this wave's 16 queries (K-dim = HD = 128 -> 4 frags)
  v16bf qf[4];
#pragma unroll
  for (int kk = 0; kk < 4; ++kk)
    qf[kk] = load_a_frag(qbg + (size_t)qbase * NHD + kk * 32, NHD);

  const float scale = 0.08838834764831845f;  // 1/sqrt(128)
  for (int kt = 0; kt < NS / 16; ++kt) {
    v8f c = vzero8();
#pragma unroll
    for (int kk = 0; kk < 4; ++kk) {
      v16bf b = load_b_frag(Ks + (kt * 16) * KLD + kk * 32, KLD);
      c = WMMA_BF16(qf[kk], b, c);
    }
    int col = kt * 16 + m;
#pragma unroll
    for (int r = 0; r < 8; ++r) {
      int row = r + 8 * half;
      float s = c[r] * scale;
      if (col > qbase + row) s = -1.0e30f;  // causal mask
      myS[row * SLD + col] = s;
    }
  }
  __syncthreads();

  // softmax: each lane handles half of one row (128 cols), combine via shfl_xor(16)
  {
    int row = m;
    int c0 = half * 128;
    float mx = -3.0e38f;
    for (int i = 0; i < 128; ++i) mx = fmaxf(mx, myS[row * SLD + c0 + i]);
    mx = fmaxf(mx, __shfl_xor(mx, 16));
    float sum = 0.0f;
    for (int i = 0; i < 128; ++i) {
      float p = __expf(myS[row * SLD + c0 + i] - mx);
      sum += p;
      myP[row * PLD + c0 + i] = (bf16_t)p;   // unnormalized probs
    }
    sum += __shfl_xor(sum, 16);
    if (half == 0) myR[row] = sum;
  }
  __syncthreads();

  // PV: ctx[16 x 128] = probs[16 x 256] * V[256 x 128], normalize by row sum
  float rinv[8];
#pragma unroll
  for (int r = 0; r < 8; ++r) rinv[r] = 1.0f / myR[r + 8 * half];

  int b = bh >> 4, h = bh & 15;
  bf16_t* cdst = ctx + ((size_t)(b * NS + qbase)) * ND + h * NHD;
  for (int ntile = 0; ntile < 8; ++ntile) {
    v8f c = vzero8();
#pragma unroll
    for (int kt = 0; kt < 8; ++kt) {
      v16bf a = load_a_frag(myP + kt * 32, PLD);
      v16bf bb = load_b_frag(Vt + (ntile * 16) * VLD + kt * 32, VLD);
      c = WMMA_BF16(a, bb, c);
    }
#pragma unroll
    for (int r = 0; r < 8; ++r)
      cdst[(size_t)(r + 8 * half) * ND + ntile * 16 + m] = (bf16_t)(c[r] * rinv[r]);
  }
}

// ---------------- launch ----------------
extern "C" void kernel_launch(void* const* d_in, const int* in_sizes, int n_in,
                              void* d_out, int out_size, void* d_ws, size_t ws_size,
                              hipStream_t stream) {
  const float* x    = (const float*)d_in[0];
  const float* wqkv = (const float*)d_in[1];
  const float* wo   = (const float*)d_in[2];
  float* out = (float*)d_out;

  char* ws = (char*)d_ws;
  size_t off = 0;
  auto carve = [&](size_t bytes) {
    void* p = ws + off;
    off += (bytes + 255) & ~(size_t)255;
    return p;
  };
  bf16_t* xb  = (bf16_t*)carve((size_t)NROWS * ND * 2);
  bf16_t* wqb = (bf16_t*)carve((size_t)N3D * ND * 2);
  bf16_t* wob = (bf16_t*)carve((size_t)ND * ND * 2);
  bf16_t* qws = (bf16_t*)carve((size_t)NB * NH * NS * NHD * 2);
  bf16_t* kws = (bf16_t*)carve((size_t)NB * NH * NS * NHD * 2);
  bf16_t* vtw = (bf16_t*)carve((size_t)NB * NH * NHD * NS * 2);
  bf16_t* ctx = (bf16_t*)carve((size_t)NROWS * ND * 2);

  // fp32 -> bf16
  {
    int n4 = NROWS * ND / 4;
    cvt_bf16_kernel<<<n4 / 256, 256, 0, stream>>>(x, xb, n4);
  }
  {
    int n4 = N3D * ND / 4;
    cvt_bf16_kernel<<<n4 / 256, 256, 0, stream>>>(wqkv, wqb, n4);
  }
  {
    int n4 = ND * ND / 4;
    cvt_bf16_kernel<<<n4 / 256, 256, 0, stream>>>(wo, wob, n4);
  }

  // QKV projection: 128x48 block tiles of 128x128
  qkv_gemm_kernel<<<(NROWS / 128) * (N3D / 128), 256, GEMM_SMEM, stream>>>(
      xb, wqb, qws, kws, vtw);

  // attention: one block per (b, h, 64-query slab)
  attn_kernel<<<NB * NH * (NS / 64), 128, ATTN_SMEM, stream>>>(qws, kws, vtw, ctx);

  // output projection: 128x16 block tiles of 128x128
  out_gemm_kernel<<<(NROWS / 128) * (ND / 128), 256, GEMM_SMEM, stream>>>(ctx, wob, out);
}
